// EffAttentionBox_31688268709938
// MI455X (gfx1250) — matrix operand order repaired and checked by reading, not compiled, not run on hardware
//
#include <hip/hip_runtime.h>
#include <hip/hip_bf16.h>

// ---------------------------------------------------------------------------
// EffAttentionBox on gfx1250 (MI455X), bf16 WMMA path with fp32 accumulation.
//   B=4, N=4096, IN_CH=1280, C=256, H=8, hd=32, CUT_RATE=4, blk=1024
// Double-buffered LDS staging, global prefetch, softmax row-sums via WMMA.
// ---------------------------------------------------------------------------

typedef __attribute__((ext_vector_type(16))) __bf16 v16bf;
typedef __attribute__((ext_vector_type(8)))  float  v8f;
typedef unsigned short u16;

#define SCALE_QK 0.07905694150420949f   // (1280//8)^-0.5

union Frag16 {
    uint4  u[2];
    v16bf  v;
};

// native f32 -> bf16 convert (RNE), lowers to hardware cvt on gfx1250
__device__ __forceinline__ u16 f32_to_bf16(float f) {
    __bf16 h = (__bf16)f;
    return __builtin_bit_cast(u16, h);
}

// A fragment: 16(M) x 32(K) bf16, row-major, rowStrideBytes multiple of 16.
// ISA layout: lanes 0-15 hold K={0..7,16..23}, lanes 16-31 hold K={8..15,24..31}
__device__ __forceinline__ v16bf load_a_frag(const u16* base, int rowStrideBytes, int lane) {
    int m = lane & 15, half = lane >> 4;
    const char* p = (const char*)base + m * rowStrideBytes + 16 * half;
    Frag16 f;
    f.u[0] = *(const uint4*)(p);        // VGPR0-3 : K = 8*half + 0..7
    f.u[1] = *(const uint4*)(p + 32);   // VGPR4-7 : K = 16 + 8*half + 0..7
    return f.v;
}

// B fragment: 32(K) x 16(N) bf16 stored N-major ([n][k]), rowStrideBytes mult of 16.
// ISA layout: lane&15 = N; lanes 0-15 K={0..7,16..23}, lanes 16-31 K={8..15,24..31}
__device__ __forceinline__ v16bf load_b_frag(const u16* base, int rowStrideBytes, int lane) {
    int n = lane & 15, half = lane >> 4;
    const char* p = (const char*)base + n * rowStrideBytes + 16 * half;
    Frag16 f;
    f.u[0] = *(const uint4*)(p);        // VGPR0-3 : K = 8*half + 0..7
    f.u[1] = *(const uint4*)(p + 32);   // VGPR4-7 : K = 16 + 8*half + 0..7
    return f.v;
}

__device__ __forceinline__ v8f wmma_bf16(v16bf a, v16bf b, v8f c) {
    return __builtin_amdgcn_wmma_f32_16x16x32_bf16(
        /*neg_a=*/false, a, /*neg_b=*/false, b,
        /*c_mod=*/(short)0, c, /*reuse_a=*/false, /*reuse_b=*/false);
}

// ---------------------------------------------------------------------------
// Weight transpose + fp32->bf16 convert:  W[K][N] (row-major f32) -> WT[N][K] bf16
//   grid: (N/256, K); coalesced reads, no integer division.
// ---------------------------------------------------------------------------
__global__ __launch_bounds__(256) void transpose_convert_kernel(
    const float* __restrict__ W, u16* __restrict__ WT, int K, int N) {
    int n = blockIdx.x * 256 + threadIdx.x;
    int k = blockIdx.y;
    WT[(size_t)n * K + k] = f32_to_bf16(W[(size_t)k * N + n]);
}

// ---------------------------------------------------------------------------
// Tiled GEMM: C[M][N] = A[M][K] @ W  where WT[N][K] is the bf16-transposed W.
//   workgroup tile: 128(M) x 64(N); 8 waves, each 16(M) x 64(N); K-step 32.
//   Double-buffered LDS: stage step ki+1 while WMMAs consume step ki.
// ---------------------------------------------------------------------------
template <bool A_BF16, bool BIAS_F32OUT>
__global__ __launch_bounds__(256) void gemm_wmma_kernel(
    const void* __restrict__ Ap, const u16* __restrict__ WT,
    const float* __restrict__ bias, void* __restrict__ Cp,
    int M, int N, int K) {
    __shared__ __align__(16) u16 lA[2][128 * 40];   // row stride 40 u16 = 80B
    __shared__ __align__(16) u16 lB[2][64 * 40];

    const int tid  = threadIdx.x;
    const int lane = tid & 31;
    const int wave = tid >> 5;
    const int bm = blockIdx.y * 128;
    const int bn = blockIdx.x * 64;

    // staging thread mapping
    const int arow = tid >> 1, akh = (tid & 1) * 16;     // A: 128 rows x 2 K-halves
    const int brow = tid >> 2, bkh = (tid & 3) * 8;      // B: 64 rows x 4 K-quarters

    auto stage = [&](int k0, int p) {
        if (A_BF16) {
            const u16* src = (const u16*)Ap + (size_t)(bm + arow) * K + k0 + akh;
            *(uint4*)&lA[p][arow * 40 + akh]     = *(const uint4*)(src);
            *(uint4*)&lA[p][arow * 40 + akh + 8] = *(const uint4*)(src + 8);
        } else {
            const float* src = (const float*)Ap + (size_t)(bm + arow) * K + k0 + akh;
            const float4* s4 = (const float4*)src;
            float4 a0 = s4[0], a1 = s4[1], a2 = s4[2], a3 = s4[3];
            u16 t[16];
            t[0]  = f32_to_bf16(a0.x); t[1]  = f32_to_bf16(a0.y);
            t[2]  = f32_to_bf16(a0.z); t[3]  = f32_to_bf16(a0.w);
            t[4]  = f32_to_bf16(a1.x); t[5]  = f32_to_bf16(a1.y);
            t[6]  = f32_to_bf16(a1.z); t[7]  = f32_to_bf16(a1.w);
            t[8]  = f32_to_bf16(a2.x); t[9]  = f32_to_bf16(a2.y);
            t[10] = f32_to_bf16(a2.z); t[11] = f32_to_bf16(a2.w);
            t[12] = f32_to_bf16(a3.x); t[13] = f32_to_bf16(a3.y);
            t[14] = f32_to_bf16(a3.z); t[15] = f32_to_bf16(a3.w);
            *(uint4*)&lA[p][arow * 40 + akh]     = *(const uint4*)&t[0];
            *(uint4*)&lA[p][arow * 40 + akh + 8] = *(const uint4*)&t[8];
        }
        const u16* srcB = WT + (size_t)(bn + brow) * K + k0 + bkh;
        *(uint4*)&lB[p][brow * 40 + bkh] = *(const uint4*)(srcB);
    };

    v8f acc[4];
#pragma unroll
    for (int nt = 0; nt < 4; ++nt)
#pragma unroll
        for (int i = 0; i < 8; ++i) acc[nt][i] = 0.0f;

    const int nk = K >> 5;   // K / 32
    stage(0, 0);
    __syncthreads();

    for (int ki = 0; ki < nk; ++ki) {
        const int p = ki & 1;
        if (ki + 1 < nk) stage((ki + 1) << 5, p ^ 1);
        if (ki + 2 < nk) {   // pull step ki+2 toward the caches (global_prefetch_b8)
            if (A_BF16)
                __builtin_prefetch((const u16*)Ap + (size_t)(bm + arow) * K + ((ki + 2) << 5) + akh, 0, 3);
            else
                __builtin_prefetch((const float*)Ap + (size_t)(bm + arow) * K + ((ki + 2) << 5) + akh, 0, 3);
            __builtin_prefetch(WT + (size_t)(bn + brow) * K + ((ki + 2) << 5) + bkh, 0, 3);
        }

        v16bf af = load_a_frag(&lA[p][(wave * 16) * 40], 80, lane);
#pragma unroll
        for (int nt = 0; nt < 4; ++nt) {
            v16bf bf = load_b_frag(&lB[p][(nt * 16) * 40], 80, lane);
            acc[nt] = wmma_bf16(af, bf, acc[nt]);
        }
        __syncthreads();
    }

    // ---- epilogue: C layout M = r + 8*(lane>=16), N = lane&15 ----
    const int n0 = lane & 15, half = lane >> 4;
#pragma unroll
    for (int nt = 0; nt < 4; ++nt) {
#pragma unroll
        for (int r = 0; r < 8; ++r) {
            int m = bm + wave * 16 + r + 8 * half;
            int n = bn + nt * 16 + n0;
            if (BIAS_F32OUT) {
                ((float*)Cp)[(size_t)m * N + n] = acc[nt][r] + bias[n];
            } else {
                ((u16*)Cp)[(size_t)m * N + n] = f32_to_bf16(acc[nt][r]);
            }
        }
    }
}

// ---------------------------------------------------------------------------
// Block-diagonal attention, flash-style softmax with tile-wide running max.
//   qkv bf16 [16384][768]: cols [h*32..) = Q, [256+h*32..) = K, [512+h*32..) = V
//   grid: 1024 blocks = (b:4)(h:8)(chunk:4)(qblock:8); 8 waves x 16 queries.
//   Per key-tile of 32: S = 2 WMMA, exp, row-sums via WMMA vs ones-matrix
//   (C-layout of the sum matches runsum registers), O += P@V = 2 WMMA.
// ---------------------------------------------------------------------------
__global__ __launch_bounds__(256) void attn_wmma_kernel(
    const u16* __restrict__ qkv, u16* __restrict__ o) {
    __shared__ __align__(16) u16 lK [2][128 * 32];  // [key][hd]  = B-frag layout for QK^T
    __shared__ __align__(16) u16 lVT[2][32 * 128];  // [hd][key]  = B-frag layout for P@V
    __shared__ __align__(16) u16 lP [8 * 16 * 32];  // per-wave P scratch (C->A relayout)

    const int tid  = threadIdx.x;
    const int lane = tid & 31;
    const int wave = tid >> 5;

    int bi = blockIdx.x;
    const int qblock = bi & 7;  bi >>= 3;
    const int chunk  = bi & 3;  bi >>= 2;
    const int h      = bi & 7;  bi >>= 3;
    const int b      = bi;
    const int T0   = b * 4096 + chunk * 1024;         // chunk token base
    const int qrow = T0 + qblock * 128 + wave * 16;   // this wave's first query token

    const int skey = tid >> 1, sdh = (tid & 1) * 16;  // staging map: 128 keys x 2 halves

    auto stageKV = [&](int kb, int p) {
        // K superblock: lK[p][key][hd]
        const u16* srcK = qkv + (size_t)(T0 + kb * 128 + skey) * 768 + 256 + h * 32 + sdh;
        *(uint4*)&lK[p][skey * 32 + sdh]     = *(const uint4*)(srcK);
        *(uint4*)&lK[p][skey * 32 + sdh + 8] = *(const uint4*)(srcK + 8);
        // V transposed: lVT[p][hd][key]
        const u16* srcV = qkv + (size_t)(T0 + kb * 128 + skey) * 768 + 512 + h * 32 + sdh;
        u16 t[16];
        *(uint4*)&t[0] = *(const uint4*)(srcV);
        *(uint4*)&t[8] = *(const uint4*)(srcV + 8);
#pragma unroll
        for (int j = 0; j < 16; ++j) lVT[p][(sdh + j) * 128 + skey] = t[j];
    };

    // ones B-fragment (bf16 1.0 splat) for row-sum WMMA
    v16bf ones;
    {
        Frag16 f;
        f.u[0] = uint4{0x3F803F80u, 0x3F803F80u, 0x3F803F80u, 0x3F803F80u};
        f.u[1] = f.u[0];
        ones = f.v;
    }

    // Q fragment (A layout) straight from global: row m at qkv[(qrow+m)*768 + h*32]
    v16bf qf;
    {
        int m = lane & 15, half = lane >> 4;
        const char* p = (const char*)(qkv + (size_t)(qrow + m) * 768 + h * 32) + 16 * half;
        Frag16 f;
        f.u[0] = *(const uint4*)(p);
        f.u[1] = *(const uint4*)(p + 32);
        qf = f.v;
    }

    v8f o0, o1;
    float runsum[8];
    float rm = -3.0e38f;                 // tile-wide running max (scaled domain)
#pragma unroll
    for (int r = 0; r < 8; ++r) {
        o0[r] = 0.0f; o1[r] = 0.0f; runsum[r] = 0.0f;
    }

    const int cc = lane & 15, chalf = lane >> 4;
    u16* pbase = &lP[wave * 16 * 32];

    stageKV(0, 0);
    __syncthreads();

    for (int kb = 0; kb < 8; ++kb) {             // 8 superblocks x 128 keys = 1024
        const int p = kb & 1;
        if (kb + 1 < 8) stageKV(kb + 1, p ^ 1);
        if (kb + 2 < 8) {                        // prefetch superblock kb+2
            __builtin_prefetch(qkv + (size_t)(T0 + (kb + 2) * 128 + skey) * 768 + 256 + h * 32 + sdh, 0, 3);
            __builtin_prefetch(qkv + (size_t)(T0 + (kb + 2) * 128 + skey) * 768 + 512 + h * 32 + sdh, 0, 3);
        }

#pragma unroll
        for (int jt = 0; jt < 4; ++jt) {         // 4 key-tiles of 32 per superblock
            // S = Q @ K^T  (two 16x16 n-tiles over keys), raw (unscaled)
            v16bf bk0 = load_b_frag(&lK[p][(jt * 32) * 32],      64, lane);
            v16bf bk1 = load_b_frag(&lK[p][(jt * 32 + 16) * 32], 64, lane);
            v8f z;
#pragma unroll
            for (int i = 0; i < 8; ++i) z[i] = 0.0f;
            v8f s0 = wmma_bf16(qf, bk0, z);
            v8f s1 = wmma_bf16(qf, bk1, z);

            // tile-wide max (safe upper bound for every row; shift cancels in softmax)
            float tm = fmaxf(s0[0], s1[0]);
#pragma unroll
            for (int r = 1; r < 8; ++r) tm = fmaxf(tm, fmaxf(s0[r], s1[r]));
#pragma unroll
            for (int off = 1; off < 32; off <<= 1)
                tm = fmaxf(tm, __shfl_xor(tm, off));
            const float nm = fmaxf(rm, tm * SCALE_QK);
            const float sc = __expf(rm - nm);    // rescale factor for old state
            rm = nm;

            // P = exp(S*scale - nm), written to per-wave LDS in C-layout
#pragma unroll
            for (int r = 0; r < 8; ++r) {
                float e0 = __expf(fmaf(s0[r], SCALE_QK, -nm));
                float e1 = __expf(fmaf(s1[r], SCALE_QK, -nm));
                int m = r + 8 * chalf;
                pbase[m * 32 + cc]      = f32_to_bf16(e0);
                pbase[m * 32 + 16 + cc] = f32_to_bf16(e1);
                o0[r] *= sc;
                o1[r] *= sc;
            }

            // reload P as A-fragment (LDS ops in-order per wave)
            v16bf pf = load_a_frag(pbase, 64, lane);

            // row-sums of P via WMMA against ones: C-layout matches runsum[r]
            v8f rs = wmma_bf16(pf, ones, z);
#pragma unroll
            for (int r = 0; r < 8; ++r) runsum[r] = runsum[r] * sc + rs[r];

            // O += P @ V
            v16bf bv0 = load_b_frag(&lVT[p][0 * 128  + jt * 32], 256, lane);
            v16bf bv1 = load_b_frag(&lVT[p][16 * 128 + jt * 32], 256, lane);
            o0 = wmma_bf16(pf, bv0, o0);
            o1 = wmma_bf16(pf, bv1, o1);
        }
        __syncthreads();
    }

    // ---- epilogue: o[token][h*32 + d], normalize by running sum ----
    {
#pragma unroll
        for (int r = 0; r < 8; ++r) {
            int m = r + 8 * chalf;
            float inv = 1.0f / runsum[r];
            size_t row = (size_t)(qrow + m) * 256 + h * 32;
            o[row + cc]      = f32_to_bf16(o0[r] * inv);
            o[row + 16 + cc] = f32_to_bf16(o1[r] * inv);
        }
    }
}

// ---------------------------------------------------------------------------
// Host-side orchestration
// ---------------------------------------------------------------------------
extern "C" void kernel_launch(void* const* d_in, const int* in_sizes, int n_in,
                              void* d_out, int out_size, void* d_ws, size_t ws_size,
                              hipStream_t stream) {
    const float* x    = (const float*)d_in[0];   // [4,4096,1280]
    const float* Wd   = (const float*)d_in[1];   // [1280,256]
    const float* Wqkv = (const float*)d_in[2];   // [256,768]
    const float* Wup  = (const float*)d_in[3];   // [256,1280]
    const float* bup  = (const float*)d_in[4];   // [1280]
    float* out = (float*)d_out;                  // [4,4096,1280]

    const int M = 4 * 4096;                      // 16384 tokens

    // workspace carve-up (all 16B aligned)
    char* ws = (char*)d_ws;
    u16* WdT   = (u16*)(ws);                                  //  256 x 1280
    u16* WqkvT = (u16*)(ws + 655360);                         //  768 x  256
    u16* WupT  = (u16*)(ws + 655360 + 393216);                // 1280 x  256
    u16* hbuf  = (u16*)(ws + 1703936);                        // 16384 x 256
    u16* qkv   = (u16*)(ws + 1703936 + 8388608);              // 16384 x 768
    u16* obuf  = (u16*)(ws + 1703936 + 8388608 + 25165824);   // 16384 x 256

    // 1) weights -> bf16, transposed to [N][K]  (grid: N/256 x K)
    transpose_convert_kernel<<<dim3(256 / 256, 1280), 256, 0, stream>>>(Wd,   WdT,   1280, 256);
    transpose_convert_kernel<<<dim3(768 / 256,  256), 256, 0, stream>>>(Wqkv, WqkvT,  256, 768);
    transpose_convert_kernel<<<dim3(1280 / 256, 256), 256, 0, stream>>>(Wup,  WupT,   256, 1280);

    // 2) h = x @ Wd          [16384,1280]x[1280,256] -> bf16
    gemm_wmma_kernel<false, false><<<dim3(256 / 64, M / 128), 256, 0, stream>>>(
        (const void*)x, WdT, nullptr, (void*)hbuf, M, 256, 1280);

    // 3) qkv = h @ Wqkv      [16384,256]x[256,768] -> bf16
    gemm_wmma_kernel<true, false><<<dim3(768 / 64, M / 128), 256, 0, stream>>>(
        (const void*)hbuf, WqkvT, nullptr, (void*)qkv, M, 768, 256);

    // 4) block-diagonal attention -> obuf bf16 [16384,256]
    attn_wmma_kernel<<<1024, 256, 0, stream>>>(qkv, obuf);

    // 5) out = o @ Wup + bup [16384,256]x[256,1280] -> f32
    gemm_wmma_kernel<true, true><<<dim3(1280 / 64, M / 128), 256, 0, stream>>>(
        (const void*)obuf, WupT, bup, (void*)out, M, 1280, 256);
}
